// RadNet_24713241822283
// MI455X (gfx1250) — matrix operand order repaired and checked by reading, not compiled, not run on hardware
//
#include <hip/hip_runtime.h>
#include <hip/hip_bf16.h>

typedef __attribute__((ext_vector_type(16))) _Float16 v16h;
typedef __attribute__((ext_vector_type(8)))  _Float16 v8h;
typedef __attribute__((ext_vector_type(8)))  float    v8f;
typedef __attribute__((ext_vector_type(4)))  unsigned int u32x4;
typedef __attribute__((ext_vector_type(8)))  int          i32x8;
typedef __attribute__((ext_vector_type(4)))  int          i32x4;

#if defined(__has_builtin)
#if __has_builtin(__builtin_amdgcn_tensor_load_to_lds) && \
    __has_builtin(__builtin_amdgcn_s_wait_tensorcnt)
#define HAVE_TDM 1
#endif
#endif
#ifndef HAVE_TDM
#define HAVE_TDM 0
#endif

// ---------------------------------------------------------------- utilities
__device__ __forceinline__ float elu_f(float x) {
    return x > 0.f ? x : (__expf(x) - 1.f);
}

#if HAVE_TDM
// 1-D contiguous global -> LDS copy through the Tensor Data Mover.
// elems = number of 2-byte elements (must be <= 65535).
__device__ __forceinline__ void tdm_copy_1d(const void* gsrc, void* lds_dst,
                                            unsigned elems) {
    unsigned long long ga = (unsigned long long)gsrc;
    unsigned la = (unsigned)(unsigned long long)lds_dst;  // LDS byte offset
    u32x4 g0;
    g0[0] = 1u;                                   // count=1, user descriptor
    g0[1] = la;                                   // lds_addr
    g0[2] = (unsigned)ga;                         // global_addr[31:0]
    g0[3] = (unsigned)(ga >> 32) | (2u << 30);    // global_addr[56:32] | type=2
    i32x8 g1;
    g1[0] = (1 << 16);                            // wg_mask=0, data_size=1 (2B)
    g1[1] = (int)((elems & 0xFFFFu) << 16);       // tensor_dim0[15:0]
    g1[2] = (int)(elems >> 16);                   // tensor_dim0[31:16]
    g1[3] = (int)(elems << 16);                   // tile_dim0
    g1[4] = 0;                                    // tile_dim1=0, tile_dim2=0
    g1[5] = (int)elems;                           // tensor_dim0_stride[31:0]
    g1[6] = 0;
    g1[7] = 0;
    i32x4 z4 = {0, 0, 0, 0};
    i32x8 z8 = {0, 0, 0, 0, 0, 0, 0, 0};
    // 6-arg form (amdgpu-toolchain / clang-23 signature)
    __builtin_amdgcn_tensor_load_to_lds(g0, g1, z4, z4, z8, 0);
}
#endif

__global__ void cvt_f32_to_f16(const float* __restrict__ src,
                               _Float16* __restrict__ dst, int n) {
    int i = blockIdx.x * blockDim.x + threadIdx.x;
    if (i < n) dst[i] = (_Float16)src[i];
}

// fc1 weights: [1024][4000] f32 -> transposed [4000][1024] f16
__global__ void cvt_transpose_fc1(const float* __restrict__ src,
                                  _Float16* __restrict__ dst,
                                  int rows, int cols) {
    int idx = blockIdx.x * blockDim.x + threadIdx.x;
    if (idx >= rows * cols) return;
    int j = idx / cols, k = idx - j * cols;
    dst[(size_t)k * rows + j] = (_Float16)src[idx];
}

// ---------------------------------------------------------------- embedding
__global__ void embed_kernel(const float* __restrict__ pos,
                             const float* __restrict__ Z,
                             const float* __restrict__ neigh,
                             const int*   __restrict__ useN,
                             _Float16* __restrict__ x0, int N, int M) {
    int idx = blockIdx.x * blockDim.x + threadIdx.x;
    if (idx >= N * 729) return;
    int n = idx / 729, g = idx - n * 729;
    const float step = 2.0f * 0.926f / 8.0f;
    float gx = -0.926f + (float)(g / 81)      * step;
    float gy = -0.926f + (float)((g / 9) % 9) * step;
    float gz = -0.926f + (float)(g % 9)       * step;
    float px = pos[n * 3 + 0] - gx;
    float py = pos[n * 3 + 1] - gy;
    float pz = pos[n * 3 + 2] - gz;
    float s = 0.f;
    for (int m = 0; m < M; ++m) {
        float dx = px - neigh[((size_t)n * M + m) * 3 + 0];
        float dy = py - neigh[((size_t)n * M + m) * 3 + 1];
        float dz = pz - neigh[((size_t)n * M + m) * 3 + 2];
        float d2 = dx * dx + dy * dy + dz * dz;
        float msk = (float)useN[(size_t)n * M + m];
        s += msk * __expf(-2.0f * d2);            // -0.5/sigma^2 = -2
    }
    x0[idx] = (_Float16)(Z[n] * s);
}

// ---------------------------------------------------------------- conv3d (implicit GEMM via WMMA)
// in : [N][Cin][Din^3] f16, out: [N][Cout][Dout^3] f16
// LDS input is halo-padded [Cin][(Din+2)^3] with zero border -> branch-free gather.
// K axis is reordered to k = tap*Cin + ci (Cin power of two) -> shift/mask decode.
// K padding (k >= KK): A rows zero-filled, so garbage B reads contribute nothing.
__global__ __launch_bounds__(128)
void conv3d_wmma(const _Float16* __restrict__ in,
                 const _Float16* __restrict__ wgt,   // [Cout][Cin*27], layout ci*27+tap
                 const float*    __restrict__ bias,
                 _Float16*       __restrict__ out,
                 int Cin, int lc, int Cout, int Din, int Dout, int stride) {
    extern __shared__ _Float16 smem[];
    const int Dp  = Din + 2, Dp2 = Dp * Dp, Dp3 = Dp2 * Dp;
    const int Din2 = Din * Din, Din3 = Din2 * Din;
    const int Dout2 = Dout * Dout, Dout3 = Dout2 * Dout;
    const int KK = Cin * 27, KKp = (KK + 31) & ~31, kSteps = KKp >> 5;

    _Float16* sIn = smem;                            // Cin*Dp3 halves (halo padded)
    _Float16* sW  = smem + ((Cin * Dp3 + 7) & ~7);   // 16*KKp halves, 16B aligned

    const int n   = blockIdx.x;
    const int co0 = blockIdx.y * 16;
    const int tid = threadIdx.x;

    // pass 1: zero the padded input volume; stage reordered, zero-padded weights
    for (int i = tid; i < Cin * Dp3; i += 128) sIn[i] = (_Float16)0.f;
    for (int i = tid; i < 16 * KKp; i += 128) {
        int co = i / KKp, r = i - co * KKp;
        int tap = r >> lc, ci = r & (Cin - 1);
        sW[i] = (tap < 27) ? wgt[(size_t)(co0 + co) * KK + ci * 27 + tap]
                           : (_Float16)0.f;
    }
    __syncthreads();
    // pass 2: fill interior
    const size_t inBase = (size_t)n * Cin * Din3;
    for (int i = tid; i < Cin * Din3; i += 128) {
        int ci = i / Din3, r = i - ci * Din3;
        int z = r / Din2, r2 = r - z * Din2;
        int y = r2 / Din, x = r2 - y * Din;
        sIn[ci * Dp3 + ((z + 1) * Dp + (y + 1)) * Dp + (x + 1)] = in[inBase + i];
    }
    __syncthreads();

    const int wave = tid >> 5, lane = tid & 31;
    const int mrow = lane & 15, hi = lane >> 4;
    const int nTiles = (Dout3 + 15) >> 4;
    const int cmask = Cin - 1;

    for (int tile = wave; tile < nTiles; tile += 4) {
        // per-tile spatial bases in the padded volume (tap (0,0,0) aligns to od*stride)
        int baseT[16];
#pragma unroll
        for (int t = 0; t < 16; ++t) {
            int p = tile * 16 + t;
            if (p > Dout3 - 1) p = Dout3 - 1;        // tail: duplicate, masked at store
            int od = p / Dout2, rem = p - od * Dout2;
            int oh = rem / Dout, ow = rem - oh * Dout;
            baseT[t] = ((od * stride) * Dp + oh * stride) * Dp + ow * stride;
        }

        v8f acc = {};
        for (int ks = 0; ks < kSteps; ++ks) {
            // A fragment (weights): two contiguous 16B LDS chunks per lane
            int rowOff = mrow * KKp + ks * 32 + hi * 8;
            v8h alo = *(const v8h*)(sW + rowOff);
            v8h ahi = *(const v8h*)(sW + rowOff + 16);
            v16h a;
#pragma unroll
            for (int t = 0; t < 8; ++t) { a[t] = alo[t]; a[t + 8] = ahi[t]; }

            // B fragment: column = spatial mrow, element t = K (table order) -> (tap, ci); branch-free gather
            int colBase = baseT[mrow];
            v16h bf;
#pragma unroll
            for (int t = 0; t < 16; ++t) {
                int k = ks * 32 + (t < 8 ? hi * 8 + t : 16 + hi * 8 + (t - 8));
                int tap = k >> lc, ci = k & cmask;
                int kd = tap / 9, tr = tap - kd * 9;
                int kh = tr / 3,  kw = tr - kh * 3;
                bf[t] = sIn[ci * Dp3 + (kd * Dp + kh) * Dp + kw + colBase];
            }

            acc = __builtin_amdgcn_wmma_f32_16x16x32_f16(
                false, a, false, bf, (short)0, acc, false, false);
        }

        // D layout: acc[v] = (M = v + 8*hi = co, N = lane&15 = spatial)
        int p = tile * 16 + mrow;
        if (p < Dout3) {
#pragma unroll
            for (int v = 0; v < 8; ++v) {
                int co = co0 + v + 8 * hi;
                float val = acc[v] + bias[co];
                out[((size_t)n * Cout + co) * Dout3 + p] = (_Float16)elu_f(val);
            }
        }
    }
}

// ---------------------------------------------------------------- FC1 GEMM (2048x1024, K=4000) via WMMA
__global__ __launch_bounds__(128)
void fc1_wmma(const _Float16* __restrict__ x,    // [2048][4000] f16
              const _Float16* __restrict__ wT,   // [4000][1024] f16 (pre-transposed)
              const float*    __restrict__ b,
              float*          __restrict__ h,    // [2048][1024] f32
              int K, int Nout) {
    extern __shared__ _Float16 sX[];             // 16 * K halves (contiguous tile)
    const int tid = threadIdx.x;
    const int m0 = blockIdx.x * 16;
#if HAVE_TDM
    if (tid < 32) {                              // wave 0 issues the TDM DMA
        tdm_copy_1d(x + (size_t)m0 * K, sX, (unsigned)(16 * K));
        __builtin_amdgcn_s_wait_tensorcnt(0);
    }
#else
    for (int i = tid; i < 16 * K; i += 128)
        sX[i] = x[(size_t)m0 * K + i];
#endif
    __syncthreads();

    const int wave = tid >> 5, lane = tid & 31;
    const int mrow = lane & 15, hi = lane >> 4;
    const int n0 = (blockIdx.y * 4 + wave) * 16;
    const int kSteps = K >> 5;

    v8f acc = {};
    for (int ks = 0; ks < kSteps; ++ks) {
        int base = ks * 32 + hi * 8;
        v8h alo = *(const v8h*)(sX + mrow * K + base);
        v8h ahi = *(const v8h*)(sX + mrow * K + base + 16);
        v16h a;
#pragma unroll
        for (int t = 0; t < 8; ++t) { a[t] = alo[t]; a[t + 8] = ahi[t]; }

        int kk = ks * 32 + hi * 8;               // B: column n0+mrow, elements = K in table order
        const _Float16* bp = wT + (size_t)kk * Nout + n0 + mrow;
        v16h bf;
#pragma unroll
        for (int t = 0; t < 8; ++t) { bf[t] = bp[(size_t)t * Nout]; bf[t + 8] = bp[(size_t)(16 + t) * Nout]; }

        acc = __builtin_amdgcn_wmma_f32_16x16x32_f16(
            false, a, false, bf, (short)0, acc, false, false);
    }

    int col = n0 + mrow;
    float bc = b[col];
#pragma unroll
    for (int v = 0; v < 8; ++v) {
        int row = m0 + v + 8 * hi;
        h[(size_t)row * Nout + col] = elu_f(acc[v] + bc);
    }
}

// ---------------------------------------------------------------- FC2 + deterministic segment sum
__global__ void fc2_kernel(const float* __restrict__ h,
                           const float* __restrict__ w,   // [6][1024]
                           const float* __restrict__ b,
                           float* __restrict__ y) {       // [2048][6]
    int idx = blockIdx.x * blockDim.x + threadIdx.x;
    if (idx >= 2048 * 6) return;
    int n = idx / 6, o = idx - n * 6;
    const float* hr = h + (size_t)n * 1024;
    const float* wr = w + (size_t)o * 1024;
    float acc = b[o];
    for (int k = 0; k < 1024; ++k) acc += hr[k] * wr[k];
    y[idx] = acc;
}

__global__ void segsum_kernel(const float* __restrict__ y,
                              const int* __restrict__ index,
                              float* __restrict__ out) {
    int t = threadIdx.x;
    if (t >= 32 * 6) return;
    int seg = t / 6, o = t - seg * 6;
    float acc = 0.f;
    for (int n = 0; n < 2048; ++n)
        if (index[n] == seg) acc += y[n * 6 + o];
    out[t] = acc;
}

// ---------------------------------------------------------------- host
extern "C" void kernel_launch(void* const* d_in, const int* in_sizes, int n_in,
                              void* d_out, int out_size, void* d_ws, size_t ws_size,
                              hipStream_t stream) {
    (void)in_sizes; (void)n_in; (void)out_size; (void)ws_size;
    const int N = 2048, M = 48;
    static const int CIN[11]  = {1, 64, 64, 16, 16, 16, 16, 64, 32, 32, 32};
    static const int LC[11]   = {0, 6, 6, 4, 4, 4, 4, 6, 5, 5, 5};   // log2(Cin)
    static const int COUT[11] = {64, 64, 16, 16, 16, 16, 64, 32, 32, 32, 32};
    static const int STR[11]  = {1, 1, 1, 1, 1, 1, 2, 1, 1, 1, 1};
    static const int DIN[11]  = {9, 9, 9, 9, 9, 9, 9, 5, 5, 5, 5};
    static const int DOUT[11] = {9, 9, 9, 9, 9, 9, 5, 5, 5, 5, 5};

    const float* pos   = (const float*)d_in[0];
    const float* Zp    = (const float*)d_in[1];
    const float* neigh = (const float*)d_in[2];
    const int*   useN  = (const int*)d_in[3];
    /* d_in[4] = cell, unused by reference */
    const int*   index = (const int*)d_in[5];
    const float* conv_w[11];
    const float* conv_b[11];
    for (int i = 0; i < 11; ++i) conv_w[i] = (const float*)d_in[6 + i];
    for (int i = 0; i < 11; ++i) conv_b[i] = (const float*)d_in[17 + i];
    const float* fc1w = (const float*)d_in[28];
    const float* fc1b = (const float*)d_in[29];
    const float* fc2w = (const float*)d_in[30];
    const float* fc2b = (const float*)d_in[31];

    char* ws = (char*)d_ws;
    size_t off = 0;
    auto carve = [&](size_t bytes) -> char* {
        char* p = ws + off;
        off += (bytes + 255) & ~(size_t)255;
        return p;
    };
    size_t actElems = (size_t)N * 64 * 729;
    _Float16* actA = (_Float16*)carve(actElems * 2);
    _Float16* actB = (_Float16*)carve(actElems * 2);
    _Float16* wf16[11];
    for (int i = 0; i < 11; ++i)
        wf16[i] = (_Float16*)carve((size_t)COUT[i] * CIN[i] * 27 * 2);
    _Float16* fc1wT = (_Float16*)carve((size_t)4000 * 1024 * 2);
    float* hbuf = (float*)carve((size_t)N * 1024 * 4);
    float* ybuf = (float*)carve((size_t)N * 6 * 4);

    // weights -> f16 (conv: straight convert; fc1: convert + transpose)
    for (int i = 0; i < 11; ++i) {
        int n = COUT[i] * CIN[i] * 27;
        cvt_f32_to_f16<<<(n + 255) / 256, 256, 0, stream>>>(conv_w[i], wf16[i], n);
    }
    {
        int n = 1024 * 4000;
        cvt_transpose_fc1<<<(n + 255) / 256, 256, 0, stream>>>(fc1w, fc1wT, 1024, 4000);
    }

    // gaussian grid embedding -> actA (= [N][1][729] f16)
    {
        int tot = N * 729;
        embed_kernel<<<(tot + 255) / 256, 256, 0, stream>>>(pos, Zp, neigh, useN, actA, N, M);
    }

    // 11 conv layers, ping-pong activations
    _Float16* cur = actA;
    _Float16* nxt = actB;
    for (int i = 0; i < 11; ++i) {
        int Din = DIN[i], Dout = DOUT[i];
        int Dp3 = (Din + 2) * (Din + 2) * (Din + 2);
        int KKp = (CIN[i] * 27 + 31) & ~31;
        size_t ldsBytes = ((size_t)((CIN[i] * Dp3 + 7) & ~7) + (size_t)16 * KKp) * 2;
        dim3 grid(N, COUT[i] / 16);
        conv3d_wmma<<<grid, 128, ldsBytes, stream>>>(
            cur, wf16[i], conv_b[i], nxt, CIN[i], LC[i], COUT[i], Din, Dout, STR[i]);
        _Float16* t = cur; cur = nxt; nxt = t;
    }

    // FC1: [2048][4000] x [4000][1024] -> hbuf (ELU)
    {
        dim3 grid(N / 16, 1024 / 64);
        size_t lds = (size_t)16 * 4000 * 2;
        fc1_wmma<<<grid, 128, lds, stream>>>(cur, fc1wT, fc1b, hbuf, 4000, 1024);
    }

    // FC2 + deterministic segment sum
    {
        int tot = N * 6;
        fc2_kernel<<<(tot + 255) / 256, 256, 0, stream>>>(hbuf, fc2w, fc2b, ybuf);
    }
    segsum_kernel<<<1, 256, 0, stream>>>(ybuf, index, (float*)d_out);
}
